// Joiner_59906203845291
// MI455X (gfx1250) — matrix operand order repaired and checked by reading, not compiled, not run on hardware
//
#include <hip/hip_runtime.h>
#include <hip/hip_bf16.h>

// RNN-T joiner for MI455X (gfx1250, wave32, WMMA).
// enc/dec projections + fused tanh-broadcast + vocab GEMM, all via
// v_wmma_f32_16x16x32_f16 with f32 accumulation.
//
// LDS A-tiles are stored in *WMMA fragment order*: fragment (mt,ks) is a
// contiguous 1 KB block, lane L's 16 halves at byte offset lane*32. The hot
// loop then reads each A fragment as one 32-byte LDS load at a compile-time
// immediate offset (no per-iteration address VALU, no bank conflicts), and
// all four M-tile fragments are fetched before the four WMMAs so a single
// s_wait covers the whole group.

typedef __attribute__((ext_vector_type(16))) _Float16 v16h;
typedef __attribute__((ext_vector_type(4)))  _Float16 v4h;
typedef __attribute__((ext_vector_type(8)))  float    v8f;

#define NB   8      // batch
#define TT   512    // encoder time steps
#define UU   64     // decoder steps
#define JJ   512    // hidden (== E == D == J)
#define VV   500    // vocab
#define VP   512    // vocab padded to 16-multiple

// ---------------------------------------------------------------------------
// Fragment-order LDS addressing for a 64 x 512 f16 A-tile (exactly 64 KB).
//
// ISA A-layout (16-bit, 16x32): lane<16 holds K=0..7 (elems 0-7) and K=16..23
// (elems 8-15) of row m=lane; lane>=16 holds K=8..15 and K=24..31 of row
// m=lane-16. Element (m,k) of fragment (mt=m>>4, ks=k>>5), r=k&31 maps to:
//   lane = (m&15) + 16*(r>>4),  elem = r&15  (contiguous K, same as B)
// Fragment (mt,ks) lives at halves offset ((ks*4+mt)<<9); lane data at +lane*16.
// ---------------------------------------------------------------------------
__device__ __forceinline__ int frag_elem_off(int m, int k) {
    const int r    = k & 31;
    const int lane = (m & 15) + ((r >> 4) << 4);
    const int pos  = r & 15;
    return ((((k >> 5) << 2) + (m >> 4)) << 9) + (lane << 4) + pos;
}

// B fragment (32x16 f16, K x N): lane = {n = lane&15, K-half = lane>>4};
// 16 contiguous K values of row n of the row-major f16 weight matrix ->
// one 32-byte global load, no transpose needed.
__device__ __forceinline__ v16h load_b_frag(const _Float16* __restrict__ bptr,
                                            int k0) {
    return *(const v16h*)(bptr + k0);
}

__device__ __forceinline__ v8f wmma_f16(v16h a, v16h b, v8f c) {
    return __builtin_amdgcn_wmma_f32_16x16x32_f16(
        /*neg_a=*/false, a, /*neg_b=*/false, b,
        /*c_mod=*/(short)0, c, /*reuse_a=*/false, /*reuse_b=*/false);
}

// ---------------------------------------------------------------------------
// Kernel 0: convert weights to f16 once. Layout in workspace (halves):
//   [0, 256K)   W_enc  [J][E]
//   [256K,512K) W_dec  [J][D]
//   [512K,768K) W_out  [VP][J], rows 500..511 zero-padded
// ---------------------------------------------------------------------------
__global__ __launch_bounds__(256) void convert_weights_kernel(
    const float* __restrict__ We, const float* __restrict__ Wd,
    const float* __restrict__ Wo, _Float16* __restrict__ dst) {
    const int idx = blockIdx.x * 256 + threadIdx.x;   // < 786432
    float val;
    if (idx < 262144) {
        val = We[idx];
    } else if (idx < 524288) {
        val = Wd[idx - 262144];
    } else {
        const int i2 = idx - 524288;
        const int v  = i2 >> 9;
        val = (v < VV) ? Wo[i2] : 0.0f;
    }
    dst[idx] = (_Float16)val;
}

// ---------------------------------------------------------------------------
// Shared GEMM stage: 64-row f16 A-tile (fragment order, LDS) x Wh[512][512]
// f16 -> 4 x v8f accumulators per wave. 8 waves; wave w owns N-tiles
// [4w,4w+4) x all 4 M-tiles; B fragment loaded once per K-step and reused
// across 4 WMMAs. All LDS offsets are immediates after unrolling.
// ---------------------------------------------------------------------------
#define GEMM_TILE_BODY(ldsTile, Wh, n0, lane, PREFETCH)                        \
    v8f acc0 = {}, acc1 = {}, acc2 = {}, acc3 = {};                            \
    {                                                                          \
        const _Float16* __restrict__ bptr =                                    \
            (Wh) + ((n0) + ((lane) & 15)) * JJ + (((lane) >> 4) << 4);         \
        const _Float16* aptr = (ldsTile) + ((lane) << 4);                      \
        _Pragma("unroll 4")                                                    \
        for (int ks = 0; ks < 16; ++ks) {                                      \
            PREFETCH;                                                          \
            const v16h b  = load_b_frag(bptr, ks << 5);                        \
            const v16h a0 = *(const v16h*)(aptr + (((ks << 2) + 0) << 9));     \
            const v16h a1 = *(const v16h*)(aptr + (((ks << 2) + 1) << 9));     \
            const v16h a2 = *(const v16h*)(aptr + (((ks << 2) + 2) << 9));     \
            const v16h a3 = *(const v16h*)(aptr + (((ks << 2) + 3) << 9));     \
            acc0 = wmma_f16(a0, b, acc0);                                      \
            acc1 = wmma_f16(a1, b, acc1);                                      \
            acc2 = wmma_f16(a2, b, acc2);                                      \
            acc3 = wmma_f16(a3, b, acc3);                                      \
        }                                                                      \
    }

// ---------------------------------------------------------------------------
// Kernel 1/2: projection GEMM  out[M][512] = A[M][512] @ W^T + bias.
// One block = 64 rows of A.
// ---------------------------------------------------------------------------
__global__ __launch_bounds__(256) void proj_gemm_kernel(
    const float* __restrict__ A, const _Float16* __restrict__ Wh,
    const float* __restrict__ bias, float* __restrict__ out) {
    __shared__ alignas(64) _Float16 aLds[64 * JJ];     // 64 KB, fragment order

    const int tid = threadIdx.x;
    const int m0g = blockIdx.x * 64;

    // Stage 1: A tile -> f16 LDS in fragment order (32 float4 per thread).
    const float4* A4 = (const float4*)(A + (size_t)m0g * JJ);
    #pragma unroll 4
    for (int i = 0; i < 32; ++i) {
        const int idx4 = tid + i * 256;
        const int m  = idx4 >> 7;                      // /128 float4 per row
        const int j  = (idx4 & 127) << 2;
        const float4 v = A4[(idx4 >> 7) * 128 + (idx4 & 127)];
        v4h h;
        h.x = (_Float16)v.x; h.y = (_Float16)v.y;
        h.z = (_Float16)v.z; h.w = (_Float16)v.w;
        *(v4h*)(aLds + frag_elem_off(m, j)) = h;       // r&7 in {0,4}: 8B aligned
    }
    __syncthreads();

    const int lane = tid & 31;
    const int wv   = tid >> 5;

    for (int nti = 0; nti < 4; ++nti) {
        const int n0 = (wv * 4 + nti) << 4;
        GEMM_TILE_BODY(aLds, Wh, n0, lane, )
        const int j  = n0 + (lane & 15);
        const float bj = bias[j];
        const int mb = (lane >> 4) << 3;               // 0 or 8
        #pragma unroll
        for (int r = 0; r < 8; ++r) {
            const int mr = mb + r;
            out[(size_t)(m0g + mr)      * JJ + j] = acc0[r] + bj;
            out[(size_t)(m0g + 16 + mr) * JJ + j] = acc1[r] + bj;
            out[(size_t)(m0g + 32 + mr) * JJ + j] = acc2[r] + bj;
            out[(size_t)(m0g + 48 + mr) * JJ + j] = acc3[r] + bj;
        }
    }
}

// ---------------------------------------------------------------------------
// Kernel 3: fused tanh-broadcast + vocab GEMM.
// One block per (n,t): act[u][j] = tanh(enc[n,t,j] + dec[n,u,j]) -> f16 LDS,
// then logits[n,t,u,v] = act @ W_out_f16^T + b_out  (64 x 500, K = 512).
// Output stores (524 MB total) are the roofline floor (~22 us @ 23.3 TB/s);
// they are non-temporal so the single-use logits stream doesn't evict W_out /
// enc_proj / dec_proj from the 192 MB L2.
// ---------------------------------------------------------------------------
__global__ __launch_bounds__(256) void joiner_vocab_kernel(
    const float* __restrict__ enc_proj,   // [N*T][512]
    const float* __restrict__ dec_proj,   // [N][64][512]
    const _Float16* __restrict__ Woh,     // [512][512] (padded f16)
    const float* __restrict__ b_out,      // [500]
    float* __restrict__ out) {            // [N][T][64][500]
    __shared__ alignas(64) _Float16 act[64 * JJ];      // 64 KB, fragment order

    const int tid = threadIdx.x;
    const int ntb = blockIdx.x;                        // n*T + t
    const int n   = ntb >> 9;                          // / T

    // Stage 1: activations -> f16 LDS in fragment order.
    const float4* enc4 = (const float4*)(enc_proj + (size_t)ntb * JJ);
    const float4* dec4 = (const float4*)(dec_proj + (size_t)n * UU * JJ);
    #pragma unroll 4
    for (int i = 0; i < 32; ++i) {
        const int idx4 = tid + i * 256;
        const int u  = idx4 >> 7;
        const int j4 = idx4 & 127;
        const float4 e = enc4[j4];
        const float4 d = dec4[u * 128 + j4];
        v4h h;
        h.x = (_Float16)tanhf(e.x + d.x);
        h.y = (_Float16)tanhf(e.y + d.y);
        h.z = (_Float16)tanhf(e.z + d.z);
        h.w = (_Float16)tanhf(e.w + d.w);
        *(v4h*)(act + frag_elem_off(u, j4 << 2)) = h;
    }
    __syncthreads();

    const int lane = tid & 31;
    const int wv   = tid >> 5;
    float* outRow = out + (size_t)ntb * UU * VV;

    for (int nti = 0; nti < 4; ++nti) {
        const int n0 = (wv * 4 + nti) << 4;
        GEMM_TILE_BODY(act, Woh, n0, lane,
            if (ks + 4 < 16)   /* pull upcoming W_out K-chunk into cache */
                __builtin_prefetch((const void*)(bptr + (ks << 5) + 128), 0, 0))
        const int v = n0 + (lane & 15);
        if (v < VV) {                                  // drop padded vocab cols
            const float bv = b_out[v];
            const int mb = (lane >> 4) << 3;
            #pragma unroll
            for (int r = 0; r < 8; ++r) {
                const int mr = mb + r;
                __builtin_nontemporal_store(acc0[r] + bv,
                    &outRow[(size_t)(mr)      * VV + v]);
                __builtin_nontemporal_store(acc1[r] + bv,
                    &outRow[(size_t)(16 + mr) * VV + v]);
                __builtin_nontemporal_store(acc2[r] + bv,
                    &outRow[(size_t)(32 + mr) * VV + v]);
                __builtin_nontemporal_store(acc3[r] + bv,
                    &outRow[(size_t)(48 + mr) * VV + v]);
            }
        }
    }
}

// ---------------------------------------------------------------------------
extern "C" void kernel_launch(void* const* d_in, const int* in_sizes, int n_in,
                              void* d_out, int out_size, void* d_ws, size_t ws_size,
                              hipStream_t stream) {
    const float* encoder_out = (const float*)d_in[0];   // [8][512][512]
    const float* decoder_out = (const float*)d_in[1];   // [8][64][512]
    const float* W_enc = (const float*)d_in[2];         // [512][512]
    const float* b_enc = (const float*)d_in[3];         // [512]
    const float* W_dec = (const float*)d_in[4];         // [512][512]
    const float* b_dec = (const float*)d_in[5];         // [512]
    const float* W_out = (const float*)d_in[6];         // [500][512]
    const float* b_out = (const float*)d_in[7];         // [500]
    float* out = (float*)d_out;                         // [8][512][64][500]

    // Workspace layout: 1.5 MB f16 weights + 8 MB enc_proj + 1 MB dec_proj.
    _Float16* wh      = (_Float16*)d_ws;
    _Float16* Wenc_h  = wh;                 // 262144 halves
    _Float16* Wdec_h  = wh + 262144;
    _Float16* Wout_h  = wh + 524288;        // padded [512][512]
    float* enc_proj   = (float*)((char*)d_ws + (size_t)786432 * 2);
    float* dec_proj   = enc_proj + (size_t)4096 * 512;

    convert_weights_kernel<<<3072, 256, 0, stream>>>(W_enc, W_dec, W_out, wh);
    proj_gemm_kernel<<<64, 256, 0, stream>>>(encoder_out, Wenc_h, b_enc, enc_proj);
    proj_gemm_kernel<<<8, 256, 0, stream>>>(decoder_out, Wdec_h, b_dec, dec_proj);
    joiner_vocab_kernel<<<NB * TT, 256, 0, stream>>>(enc_proj, dec_proj,
                                                     Wout_h, b_out, out);
}
